// E59Highway_2147483648043
// MI455X (gfx1250) — matrix operand order repaired and checked by reading, not compiled, not run on hardware
//
#include <hip/hip_runtime.h>
#include <hip/hip_bf16.h>
#include <stdint.h>

#define DIM    1024
#define DINNER 1024

typedef __attribute__((ext_vector_type(16))) __bf16 v16bf;
typedef __attribute__((ext_vector_type(8)))  float  v8f;
typedef __attribute__((ext_vector_type(4)))  int    v4i;

#define AS1 __attribute__((address_space(1)))
#define AS3 __attribute__((address_space(3)))

union Frag { uint4 q[2]; v16bf v; };

#if __has_builtin(__builtin_amdgcn_global_load_async_to_lds_b128)
#define ASYNC_STAGE 1
#endif

__device__ __forceinline__ unsigned short f2bf(float x) {
  unsigned u = __float_as_uint(x);
  u += 0x7FFFu + ((u >> 16) & 1u);          // round-to-nearest-even
  return (unsigned short)(u >> 16);
}

// stage one 16B quad: global -> LDS (async DMA path on gfx1250)
__device__ __forceinline__ void stage_b128(const unsigned short* g, unsigned short* l) {
#ifdef ASYNC_STAGE
  __builtin_amdgcn_global_load_async_to_lds_b128(
      (AS1 v4i*)(g), (AS3 v4i*)(l), 0, 0);
#else
  *(uint4*)l = *(const uint4*)g;
#endif
}

template <int N>
__device__ __forceinline__ void stage_wait() {
#ifdef ASYNC_STAGE
#if __has_builtin(__builtin_amdgcn_s_wait_asynccnt)
  __builtin_amdgcn_s_wait_asynccnt(N);
#else
  asm volatile("s_wait_asynccnt %0" :: "n"(N));
#endif
#endif
}

// ---------------------------------------------------------------------------
// f32 -> bf16 conversion (x and the three weight matrices)
// ---------------------------------------------------------------------------
__global__ void f32_to_bf16_kernel(const float* __restrict__ in,
                                   unsigned short* __restrict__ out, int n) {
  int i = blockIdx.x * blockDim.x + threadIdx.x;
  if (i < n) out[i] = f2bf(in[i]);
}

// ---------------------------------------------------------------------------
// bf16 WMMA GEMM:  C[M,N] = A[M,K] * W[N,K]^T   (+ epilogue)
//   EPI 0: out = bf16(silu(c))      -> outB
//   EPI 1: out = c + bias[n] (f32)  -> outF
//   EPI 2: out = c           (f32)  -> outF
// Block tile 128x64, 8 waves (4xM, 2xN), wave tile 32x32, K stepped by 64
// (2 WMMA k-slices / stage), double-buffered LDS filled by async-to-LDS
// copies overlapped with v_wmma_f32_16x16x32_bf16 compute.
// M%128==0, N%64==0, K%64==0 assumed (holds: 16384/1024/1024).
// ---------------------------------------------------------------------------
template <int EPI>
__global__ __launch_bounds__(256)
void gemm_bf16_wmma(const unsigned short* __restrict__ A,   // [M,K] bf16
                    const unsigned short* __restrict__ W,   // [N,K] bf16
                    const float* __restrict__ bias,         // [N]  (EPI==1)
                    float* __restrict__ outF,               // [M,N] f32
                    unsigned short* __restrict__ outB,      // [M,N] bf16
                    int M, int N, int K)
{
  constexpr int KS  = 64;
  constexpr int LDT = 72;                 // row stride in halfs (64 + 8 pad)
  __shared__ unsigned short lsA[2][128 * LDT];   // 2 x 18 KB
  __shared__ unsigned short lsB[2][ 64 * LDT];   // 2 x  9 KB

  const int tid   = threadIdx.x;
  const int lane  = tid & 31;
  const int wid   = tid >> 5;
  const int g     = lane >> 4;            // lane half (wave32)
  const int nl    = lane & 15;
  const int waveM = (wid & 3) * 32;
  const int waveN = (wid >> 2) * 32;
  const int mBase = blockIdx.y * 128;
  const int nBase = blockIdx.x * 64;

  // stage a 128x64 A tile (1024 quads) + 64x64 B tile (512 quads);
  // 6 async b128 instructions per wave per stage
  auto stage = [&](int buf, int k0) {
#pragma unroll
    for (int i = 0; i < 4; ++i) {
      const int idx = tid + i * 256;                 // 0..1023
      const int row = idx >> 3, q = (idx & 7) * 8;
      stage_b128(&A[(size_t)(mBase + row) * K + k0 + q],
                 &lsA[buf][row * LDT + q]);
    }
#pragma unroll
    for (int i = 0; i < 2; ++i) {
      const int idx = tid + i * 256;                 // 0..511
      const int row = idx >> 3, q = (idx & 7) * 8;
      stage_b128(&W[(size_t)(nBase + row) * K + k0 + q],
                 &lsB[buf][row * LDT + q]);
    }
#ifndef ASYNC_STAGE
    if (k0 + KS < K) {                               // warm next slice (WGP scope)
      __builtin_prefetch(&A[(size_t)(mBase + (tid >> 3)) * K + k0 + KS], 0, 3);
      __builtin_prefetch(&W[(size_t)(nBase + (tid >> 3)) * K + k0 + KS], 0, 3);
    }
#endif
  };

  v8f acc[2][2] = {};

  stage(0, 0);
  int buf = 0;
  for (int k0 = 0; k0 < K; k0 += KS) {
    const bool more = (k0 + KS) < K;
    if (more) {
      stage(buf ^ 1, k0 + KS);   // overlap next tile's DMA with this compute
      stage_wait<6>();           // retire current buffer's 6 (older, in-order)
    } else {
      stage_wait<0>();
    }
    __syncthreads();             // all waves' current-buffer data in LDS

#pragma unroll
    for (int s = 0; s < 2; ++s) {                    // two K=32 slices
      Frag a[2], b[2];
#pragma unroll
      for (int ti = 0; ti < 2; ++ti) {
        // A 16x32: lane half 0 -> K 0..7 / 16..23 ; half 1 -> K 8..15 / 24..31
        const unsigned short* p =
            &lsA[buf][(waveM + ti * 16 + nl) * LDT + s * 32 + 8 * g];
        a[ti].q[0] = *(const uint4*)(p);
        a[ti].q[1] = *(const uint4*)(p + 16);
      }
#pragma unroll
      for (int tj = 0; tj < 2; ++tj) {
        // B 32x16: lane half 0 -> K 0..7 / 16..23 ; half 1 -> K 8..15 / 24..31
        const unsigned short* p =
            &lsB[buf][(waveN + tj * 16 + nl) * LDT + s * 32 + 8 * g];
        b[tj].q[0] = *(const uint4*)(p);
        b[tj].q[1] = *(const uint4*)(p + 16);
      }
#pragma unroll
      for (int ti = 0; ti < 2; ++ti)
#pragma unroll
        for (int tj = 0; tj < 2; ++tj)
          acc[ti][tj] = __builtin_amdgcn_wmma_f32_16x16x32_bf16(
              false, a[ti].v, false, b[tj].v, (short)0, acc[ti][tj],
              false, false);
    }
    __syncthreads();             // everyone done reading buf before overwrite
    buf ^= 1;
  }

  // epilogue: C VGPR r holds M = r + 8*g, N = nl (16x16 f32 layout)
#pragma unroll
  for (int ti = 0; ti < 2; ++ti) {
#pragma unroll
    for (int tj = 0; tj < 2; ++tj) {
      const int n = nBase + waveN + tj * 16 + nl;
#pragma unroll
      for (int r = 0; r < 8; ++r) {
        const int m = mBase + waveM + ti * 16 + r + 8 * g;
        const float c = acc[ti][tj][r];
        if (EPI == 0) {
          const float s = c / (1.0f + __expf(-c));   // silu
          outB[(size_t)m * N + n] = f2bf(s);
        } else if (EPI == 1) {
          outF[(size_t)m * N + n] = c + bias[n];
        } else {
          outF[(size_t)m * N + n] = c;
        }
      }
    }
  }
}

// ---------------------------------------------------------------------------
// Sequential recurrence: one block per batch, 1024 threads (32 wave32 waves),
// h in registers, RMSNorm via shfl_xor intra-wave + 32-entry LDS cross-wave
// reduction. Wx is L2-resident (192 MB L2; written by GEMM2 just before).
// ---------------------------------------------------------------------------
__global__ __launch_bounds__(1024)
void scan_kernel(const float* __restrict__ Wx,        // [B*T, D] f32
                 const float* __restrict__ h0,        // [B, D]
                 const float* __restrict__ log_alpha, // scalar (device)
                 unsigned short* __restrict__ outs,   // [B*T, D] bf16
                 float* __restrict__ hfin,            // [B, D]
                 int T, int D)
{
  const int b    = blockIdx.x;
  const int d    = threadIdx.x;
  const int lane = d & 31;
  const int wid  = d >> 5;
  __shared__ float red[32];

  const float alpha = __expf(*log_alpha);
  const float invD  = 1.0f / (float)D;
  float h = h0[(size_t)b * D + d];

  for (int t = 0; t < T; ++t) {
    const size_t row = ((size_t)b * T + t) * D;
    h += alpha * Wx[row + d];

    float s = h * h;
#pragma unroll
    for (int off = 16; off; off >>= 1) s += __shfl_xor(s, off, 32);
    if (lane == 0) red[wid] = s;
    __syncthreads();
    float tot = red[lane];
#pragma unroll
    for (int off = 16; off; off >>= 1) tot += __shfl_xor(tot, off, 32);

    h *= rsqrtf(tot * invD + 1e-6f);                  // RMSNorm
    const float sig = 1.0f / (1.0f + __expf(-h));
    outs[row + d] = f2bf(h * h * sig);                // h * silu(h)
    __syncthreads();                                  // protect red (WAR)
  }
  hfin[(size_t)b * D + d] = h;
}

// ---------------------------------------------------------------------------
extern "C" void kernel_launch(void* const* d_in, const int* in_sizes, int n_in,
                              void* d_out, int out_size, void* d_ws, size_t ws_size,
                              hipStream_t stream)
{
  const float* x         = (const float*)d_in[0];
  const float* h0        = (const float*)d_in[1];
  const float* W_in      = (const float*)d_in[2];
  const float* W_cell    = (const float*)d_in[3];
  const float* b_cell    = (const float*)d_in[4];
  const float* log_alpha = (const float*)d_in[5];
  const float* W_out     = (const float*)d_in[6];

  const int M = in_sizes[0] / DIM;        // B*T = 16384
  const int B = in_sizes[1] / DINNER;     // 8
  const int T = M / B;                    // 2048

  // workspace layout; x-bf16 region is reused for the scan output
  char* ws = (char*)d_ws;
  size_t off = 0;
  unsigned short* Win_bf   = (unsigned short*)(ws + off); off += (size_t)DINNER * DIM    * 2;
  unsigned short* Wcell_bf = (unsigned short*)(ws + off); off += (size_t)DINNER * DINNER * 2;
  unsigned short* Wout_bf  = (unsigned short*)(ws + off); off += (size_t)DIM    * DINNER * 2;
  unsigned short* xbf      = (unsigned short*)(ws + off);
  unsigned short* outs_bf  = xbf;  // lifetimes disjoint: xbf dead after GEMM1
  off += (size_t)M * DIM * 2;
  unsigned short* xp_bf    = (unsigned short*)(ws + off); off += (size_t)M * DINNER * 2;
  float*          Wx       = (float*)(ws + off);          off += (size_t)M * DINNER * 4;

  float* y    = (float*)d_out;                       // [M, DIM]
  float* hfin = (float*)d_out + (size_t)M * DIM;     // [B, DINNER]

  // 1) precision conversion
  const int nx = M * DIM;
  f32_to_bf16_kernel<<<(nx + 255) / 256, 256, 0, stream>>>(x, xbf, nx);
  f32_to_bf16_kernel<<<(DINNER * DIM    + 255) / 256, 256, 0, stream>>>(W_in,   Win_bf,   DINNER * DIM);
  f32_to_bf16_kernel<<<(DINNER * DINNER + 255) / 256, 256, 0, stream>>>(W_cell, Wcell_bf, DINNER * DINNER);
  f32_to_bf16_kernel<<<(DIM    * DINNER + 255) / 256, 256, 0, stream>>>(W_out,  Wout_bf,  DIM * DINNER);

  // 2) xp = silu(x @ W_in^T)  -> bf16
  dim3 grid1(DINNER / 64, M / 128);
  gemm_bf16_wmma<0><<<grid1, 256, 0, stream>>>(xbf, Win_bf, nullptr,
                                               nullptr, xp_bf, M, DINNER, DIM);
  // 3) Wx = xp @ W_cell^T + b -> f32 (stays in L2)
  gemm_bf16_wmma<1><<<grid1, 256, 0, stream>>>(xp_bf, Wcell_bf, b_cell,
                                               Wx, nullptr, M, DINNER, DINNER);
  // 4) serial recurrence + RMSNorm + self-gate
  scan_kernel<<<B, DINNER, 0, stream>>>(Wx, h0, log_alpha, outs_bf, hfin, T, DINNER);

  // 5) y = outs @ W_out^T -> f32
  dim3 grid3(DIM / 64, M / 128);
  gemm_bf16_wmma<2><<<grid3, 256, 0, stream>>>(outs_bf, Wout_bf, nullptr,
                                               y, nullptr, M, DIM, DINNER);
}
